// AfmoeDecoderLayer_75737453297755
// MI455X (gfx1250) — matrix operand order repaired and checked by reading, not compiled, not run on hardware
//
#include <hip/hip_runtime.h>
#include <math.h>

typedef __attribute__((ext_vector_type(16))) _Float16 v16h;
typedef __attribute__((ext_vector_type(8)))  _Float16 v8h;
typedef __attribute__((ext_vector_type(8)))  float    v8f;
typedef __attribute__((ext_vector_type(4)))  float    v4f;

#ifndef __has_builtin
#define __has_builtin(x) 0
#endif
#if defined(__HIP_DEVICE_COMPILE__)
#if __has_builtin(__builtin_amdgcn_tensor_load_to_lds) && \
    __has_builtin(__builtin_amdgcn_s_wait_tensorcnt)
#define USE_TDM 1
#else
#define USE_TDM 0
#endif
#else
#define USE_TDM 0
#endif

namespace {
constexpr int   kT   = 2048;
constexpr int   kHid = 2048;
constexpr int   kNH  = 16;
constexpr int   kNKV = 4;
constexpr int   kHD  = 128;
constexpr int   kQ   = kNH * kHD;              // 2048
constexpr int   kKV  = kNKV * kHD;             // 512
constexpr int   kQKV = kQ + 2 * kKV;           // 3072
constexpr int   kWin = 1024;
constexpr float kEps = 1e-5f;
constexpr int   kE   = 32;
constexpr int   kI   = 512;
constexpr float kRouteScale = 2.0f;
}

// ------------------------- WMMA helpers -------------------------------------

__device__ inline v8f wmma32(v16h a, v16h b, v8f c) {
  // D = A(16x32 f16) x B(32x16 f16) + C(16x16 f32)
  return __builtin_amdgcn_wmma_f32_16x16x32_f16(false, a, false, b, (short)0, c,
                                                false, false);
}

__device__ inline v16h cat8(v8h lo, v8h hi) {
  return __builtin_shufflevector(lo, hi, 0, 1, 2, 3, 4, 5, 6, 7, 8, 9, 10, 11,
                                 12, 13, 14, 15);
}

__device__ inline v16h pack16(v4f a, v4f b, v4f c, v4f d) {
  v16h r;
  r[0]=(_Float16)a[0];  r[1]=(_Float16)a[1];  r[2]=(_Float16)a[2];  r[3]=(_Float16)a[3];
  r[4]=(_Float16)b[0];  r[5]=(_Float16)b[1];  r[6]=(_Float16)b[2];  r[7]=(_Float16)b[3];
  r[8]=(_Float16)c[0];  r[9]=(_Float16)c[1];  r[10]=(_Float16)c[2]; r[11]=(_Float16)c[3];
  r[12]=(_Float16)d[0]; r[13]=(_Float16)d[1]; r[14]=(_Float16)d[2]; r[15]=(_Float16)d[3];
  return r;
}

// A-matrix fragment (rows r0..r0+15, k0..k0+31), fp32 source, row-major ld.
// A layout: lanes 0-15 -> K {0..15}, lanes 16-31 -> K {16..31}.
__device__ inline v16h fragA_f32(const float* __restrict__ base, int ld, int r0,
                                 int k0, int lane) {
  const float* p = base + (size_t)(r0 + (lane & 15)) * ld + k0 + ((lane >> 4) << 4);
  v4f x0 = *(const v4f*)(p);
  v4f x1 = *(const v4f*)(p + 4);
  v4f x2 = *(const v4f*)(p + 8);
  v4f x3 = *(const v4f*)(p + 12);
  return pack16(x0, x1, x2, x3);
}

// A fragment from f16 memory (global or LDS), same chunked layout.
__device__ inline v16h fragA_f16(const _Float16* p /* row base + k0 + (lane>>4)*16 */) {
  v8h lo = *(const v8h*)(p);
  v8h hi = *(const v8h*)(p + 8);
  return cat8(lo, hi);
}

// B-matrix fragment (KxN), source stored [N,K] row-major (ld = K).
// B layout used here (same k order as A): lane n (0-15) holds column n K=0..15; lane n+16 holds K=16..31.
__device__ inline v16h fragB_f32_rowK(const float* __restrict__ base, int ld,
                                      int n0, int k0, int lane) {
  const float* p = base + (size_t)(n0 + (lane & 15)) * ld + k0 + ((lane >> 4) << 4);
  v4f x0 = *(const v4f*)(p);
  v4f x1 = *(const v4f*)(p + 4);
  v4f x2 = *(const v4f*)(p + 8);
  v4f x3 = *(const v4f*)(p + 12);
  return pack16(x0, x1, x2, x3);
}

// B fragment, source stored [K][N] (k strided, ld = N) -- global fallback.
__device__ inline v16h fragB_f32_colK(const float* __restrict__ base, int ld,
                                      int n0, int k0, int lane) {
  const float* p = base + (size_t)(k0 + ((lane >> 4) << 4)) * ld + n0 + (lane & 15);
  v16h r;
#pragma unroll
  for (int i = 0; i < 16; ++i) r[i] = (_Float16)p[(size_t)i * ld];
  return r;
}

// B fragment from f16 memory, 16 contiguous halves at p.
__device__ inline v16h fragB_f16(const _Float16* p) {
  v8h lo = *(const v8h*)(p);
  v8h hi = *(const v8h*)(p + 8);
  return cat8(lo, hi);
}

// B fragment from an LDS-staged 32x16 fp32 tile (row-major, 16 cols).
__device__ inline v16h fragB_ldsTile(const float* t, int lane) {
  const float* p = t + (((lane >> 4) << 4) * 16) + (lane & 15);
  v16h r;
#pragma unroll
  for (int i = 0; i < 16; ++i) r[i] = (_Float16)p[i * 16];
  return r;
}

// ------------------------- Tensor Data Mover --------------------------------

#if USE_TDM
typedef __attribute__((ext_vector_type(4))) unsigned int u32x4;
typedef __attribute__((ext_vector_type(8))) int          i32x8;
typedef __attribute__((ext_vector_type(4))) int          i32x4;

// 2D tile DMA: gather tile_rows x tile_cols fp32 elements (row stride
// row_stride_elems) from global into a compact LDS tile.  D# bitfields per
// CDNA5 ISA section 8.3/8.4 (group0: count/lds_addr/global_addr/type,
// group1: data_size/dims/tile dims/stride).  Tracked by TENSORcnt.
__device__ inline void tdm_load_tile_f32(unsigned lds_off,
                                         const float* __restrict__ gsrc,
                                         int tile_cols, int tile_rows,
                                         int row_stride_elems, int tensor_cols,
                                         int tensor_rows) {
  unsigned long long ga = (unsigned long long)(size_t)gsrc;
  u32x4 g0;
  g0[0] = 1u;                                            // count=1 (valid)
  g0[1] = lds_off;                                       // lds byte address
  g0[2] = (unsigned)(ga & 0xffffffffull);                // global_addr[31:0]
  g0[3] = (unsigned)((ga >> 32) & 0x01ffffffull) | (2u << 30);  // hi | type=2
  i32x8 g1;
  g1[0] = (int)(2u << 16);                               // data_size=4B
  g1[1] = (int)(((unsigned)tensor_cols & 0xffffu) << 16);         // dim0.lo
  g1[2] = (int)(((unsigned)tensor_cols >> 16) & 0xffffu) |
          (int)(((unsigned)tensor_rows & 0xffffu) << 16);         // dim0.hi|dim1.lo
  g1[3] = (int)(((unsigned)tensor_rows >> 16) & 0xffffu) |
          (int)(((unsigned)tile_cols & 0xffffu) << 16);           // dim1.hi|tile0
  g1[4] = (int)((unsigned)tile_rows & 0xffffu);                   // tile1|tile2=0
  g1[5] = row_stride_elems;                                       // dim0_stride lo
  g1[6] = 0;
  g1[7] = 0;
  i32x4 z4 = {0, 0, 0, 0};
#if __clang_major__ >= 23
  i32x8 z8 = {0, 0, 0, 0, 0, 0, 0, 0};
  __builtin_amdgcn_tensor_load_to_lds(g0, g1, z4, z4, z8, 0);
#else
  __builtin_amdgcn_tensor_load_to_lds(g0, g1, z4, z4, 0);
#endif
}
#endif  // USE_TDM

// ------------------------- reductions ---------------------------------------

__device__ inline float waveReduceSum(float x) {
#pragma unroll
  for (int m = 16; m >= 1; m >>= 1) x += __shfl_xor(x, m, 32);
  return x;
}

__device__ inline float blockReduceSum(float x) {
  __shared__ float s[8];
  int lane = threadIdx.x & 31, wv = threadIdx.x >> 5;
  x = waveReduceSum(x);
  if (lane == 0) s[wv] = x;
  __syncthreads();
  float r = (lane < (int)(blockDim.x >> 5)) ? s[lane] : 0.f;
  r = waveReduceSum(r);
  __syncthreads();  // make helper reusable within one kernel
  return r;
}

// ------------------------- kernels ------------------------------------------

// residual1 = hs + res_in ; h1 = rms(residual1) * w
__global__ __launch_bounds__(256) void add_norm_kernel(
    const float* __restrict__ hs, const float* __restrict__ res_in,
    const float* __restrict__ w, float* __restrict__ res_out,
    float* __restrict__ h_out) {
  int t = blockIdx.x, tid = threadIdx.x;
  float v[8], ss = 0.f;
#pragma unroll
  for (int i = 0; i < 8; ++i) {
    int c = tid + i * 256;
    float x = hs[(size_t)t * kHid + c] + res_in[(size_t)t * kHid + c];
    v[i] = x;
    ss += x * x;
  }
  ss = blockReduceSum(ss);
  float rs = rsqrtf(ss / kHid + kEps);
#pragma unroll
  for (int i = 0; i < 8; ++i) {
    int c = tid + i * 256;
    res_out[(size_t)t * kHid + c] = v[i];
    h_out[(size_t)t * kHid + c] = v[i] * rs * w[c];
  }
}

// C[M,N] = A[M,K] @ B[N,K]^T  (fp32 memory, fp16 WMMA math).
// Each wave owns a 16x64 strip: one A fragment feeds 4 WMMAs.
__global__ __launch_bounds__(128) void gemm_nt_kernel(
    const float* __restrict__ A, const float* __restrict__ B,
    float* __restrict__ C, int M, int N, int K) {
  int lane = threadIdx.x & 31;
  int wv = threadIdx.x >> 5;
  int n0 = (blockIdx.x * 4 + wv) * 64;
  int m0 = blockIdx.y * 16;
  if (n0 >= N) return;
  bool t1 = (n0 + 16) < N, t2 = (n0 + 32) < N, t3 = (n0 + 48) < N;
  v8f acc0 = {}, acc1 = {}, acc2 = {}, acc3 = {};
  for (int k0 = 0; k0 < K; k0 += 32) {
    if (k0 + 32 < K) {  // stream next tiles into cache: global_prefetch_b8
      __builtin_prefetch(A + (size_t)(m0 + (lane & 15)) * K + k0 + 32);
      __builtin_prefetch(B + (size_t)(n0 + (lane & 15)) * K + k0 + 32);
    }
    v16h a = fragA_f32(A, K, m0, k0, lane);
    acc0 = wmma32(a, fragB_f32_rowK(B, K, n0, k0, lane), acc0);
    if (t1) acc1 = wmma32(a, fragB_f32_rowK(B, K, n0 + 16, k0, lane), acc1);
    if (t2) acc2 = wmma32(a, fragB_f32_rowK(B, K, n0 + 32, k0, lane), acc2);
    if (t3) acc3 = wmma32(a, fragB_f32_rowK(B, K, n0 + 48, k0, lane), acc3);
  }
  int hf = lane >> 4, col = lane & 15;
#pragma unroll
  for (int r = 0; r < 8; ++r) {
    int row = m0 + r + hf * 8;
    C[(size_t)row * N + n0 + col] = acc0[r];
    if (t1) C[(size_t)row * N + n0 + 16 + col] = acc1[r];
    if (t2) C[(size_t)row * N + n0 + 32 + col] = acc2[r];
    if (t3) C[(size_t)row * N + n0 + 48 + col] = acc3[r];
  }
}

// per-head RMSNorm + RoPE on q/k, convert v -> transposed f16 [kv][d][t]
__global__ __launch_bounds__(128) void qkv_prep_kernel(
    const float* __restrict__ qkv, const int* __restrict__ positions,
    const float* __restrict__ w_qn, const float* __restrict__ w_kn,
    _Float16* __restrict__ q16, _Float16* __restrict__ k16,
    _Float16* __restrict__ vT) {
  int t = blockIdx.x;
  int wv = threadIdx.x >> 5, lane = threadIdx.x & 31;
  float pos = (float)positions[t];
  for (int job = wv; job < 24; job += 4) {
    int off, kind;
    if (job < 16)      { kind = 0; off = job * kHD; }
    else if (job < 20) { kind = 1; off = kQ + (job - 16) * kHD; }
    else               { kind = 2; off = kQ + kKV + (job - 20) * kHD; }
    const float* src = qkv + (size_t)t * kQKV + off;
    float a = src[lane], b = src[lane + 32], c = src[lane + 64], d = src[lane + 96];
    if (kind == 2) {
      int kv = job - 20;
      size_t rb = (size_t)kv * kHD * kT;
      vT[rb + (size_t)(lane)      * kT + t] = (_Float16)a;
      vT[rb + (size_t)(lane + 32) * kT + t] = (_Float16)b;
      vT[rb + (size_t)(lane + 64) * kT + t] = (_Float16)c;
      vT[rb + (size_t)(lane + 96) * kT + t] = (_Float16)d;
      continue;
    }
    float ss = waveReduceSum(a * a + b * b + c * c + d * d);
    float rs = rsqrtf(ss / kHD + kEps);
    const float* wn = (kind == 0) ? w_qn : w_kn;
    float na = a * rs * wn[lane];
    float nb = b * rs * wn[lane + 32];
    float nc = c * rs * wn[lane + 64];
    float nd = d * rs * wn[lane + 96];
    // RoPE pairs: (d, d+64) for d in [0,64)
    float invA = __powf(10000.0f, -(float)lane / 64.0f);
    float invB = __powf(10000.0f, -(float)(lane + 32) / 64.0f);
    float sA, cA, sB, cB;
    __sincosf(pos * invA, &sA, &cA);
    __sincosf(pos * invB, &sB, &cB);
    float o0 = na * cA - nc * sA;  // d = lane
    float o2 = nc * cA + na * sA;  // d = lane + 64
    float o1 = nb * cB - nd * sB;  // d = lane + 32
    float o3 = nd * cB + nb * sB;  // d = lane + 96
    _Float16* dst = (kind == 0)
        ? q16 + ((size_t)t * kNH + job) * kHD
        : k16 + ((size_t)t * kNKV + (job - 16)) * kHD;
    dst[lane]      = (_Float16)o0;
    dst[lane + 32] = (_Float16)o1;
    dst[lane + 64] = (_Float16)o2;
    dst[lane + 96] = (_Float16)o3;
  }
}

// flash-style sliding-window attention; one wave per (q-tile, head)
__global__ __launch_bounds__(32) void attn_kernel(
    const _Float16* __restrict__ q16, const _Float16* __restrict__ k16,
    const _Float16* __restrict__ vT, float* __restrict__ attn_out) {
  __shared__ _Float16 pl[16][32];  // P tile staging (D-layout -> A-layout)
  int lane = threadIdx.x;
  int hf = lane >> 4, col = lane & 15;
  int q0 = blockIdx.x * 16;
  int head = blockIdx.y;
  int kv = head >> 2;  // NH/NKV = 4

  v16h qf[4];
#pragma unroll
  for (int ks = 0; ks < 4; ++ks) {
    const _Float16* p =
        q16 + ((size_t)(q0 + col) * kNH + head) * kHD + ks * 32 + hf * 16;
    qf[ks] = fragA_f16(p);
  }

  v8f o[8] = {};
  float rmax[8], rsum[8];
#pragma unroll
  for (int r = 0; r < 8; ++r) { rmax[r] = -__builtin_inff(); rsum[r] = 0.f; }

  int lo_j = q0 - (kWin - 1);
  if (lo_j < 0) lo_j = 0;
  int kt0 = lo_j & ~31;
  const float sc = 0.08838834764831845f;  // 1/sqrt(128)

  for (int kt = kt0; kt <= q0 + 15; kt += 32) {
    v8f s0 = {}, s1 = {};
#pragma unroll
    for (int ks = 0; ks < 4; ++ks) {
      int key0 = kt + col;        if (key0 > kT - 1) key0 = kT - 1;
      int key1 = kt + 16 + col;   if (key1 > kT - 1) key1 = kT - 1;
      const _Float16* p0 =
          k16 + ((size_t)key0 * kNKV + kv) * kHD + ks * 32 + hf * 16;
      const _Float16* p1 =
          k16 + ((size_t)key1 * kNKV + kv) * kHD + ks * 32 + hf * 16;
      s0 = wmma32(qf[ks], fragB_f16(p0), s0);
      s1 = wmma32(qf[ks], fragB_f16(p1), s1);
    }
    float resc[8];
#pragma unroll
    for (int r = 0; r < 8; ++r) {
      int m = r + hf * 8;
      int i = q0 + m;
      int j0 = kt + col, j1 = j0 + 16;
      float x0 = (j0 <= i && (i - j0) < kWin) ? s0[r] * sc : -__builtin_inff();
      float x1 = (j1 <= i && (i - j1) < kWin) ? s1[r] * sc : -__builtin_inff();
      float rm = fmaxf(x0, x1);
#pragma unroll
      for (int msk = 8; msk >= 1; msk >>= 1) rm = fmaxf(rm, __shfl_xor(rm, msk, 32));
      float nm = fmaxf(rmax[r], rm);
      float rescale = (nm == -__builtin_inff()) ? 1.f : __expf(rmax[r] - nm);
      float e0 = (x0 == -__builtin_inff()) ? 0.f : __expf(x0 - nm);
      float e1 = (x1 == -__builtin_inff()) ? 0.f : __expf(x1 - nm);
      float psum = e0 + e1;
#pragma unroll
      for (int msk = 8; msk >= 1; msk >>= 1) psum += __shfl_xor(psum, msk, 32);
      rsum[r] = rsum[r] * rescale + psum;
      rmax[r] = nm;
      resc[r] = rescale;
      pl[m][col]      = (_Float16)e0;
      pl[m][col + 16] = (_Float16)e1;
    }
    __syncthreads();
#pragma unroll
    for (int j = 0; j < 8; ++j)
#pragma unroll
      for (int r = 0; r < 8; ++r) o[j][r] *= resc[r];
    v16h pa = fragA_f16(&pl[col][hf * 16]);  // ds_load path
#pragma unroll
    for (int j = 0; j < 8; ++j) {
      int d = j * 16 + col;
      const _Float16* vp = vT + ((size_t)kv * kHD + d) * kT + kt + hf * 16;
      o[j] = wmma32(pa, fragB_f16(vp), o[j]);
    }
    __syncthreads();
  }
#pragma unroll
  for (int j = 0; j < 8; ++j) {
    int d = j * 16 + col;
#pragma unroll
    for (int r = 0; r < 8; ++r) {
      int m = r + hf * 8;
      attn_out[(size_t)(q0 + m) * kQ + head * kHD + d] = o[j][r] / rsum[r];
    }
  }
}

// attn *= sigmoid(gate), in place
__global__ void gate_kernel(float* __restrict__ attn,
                            const float* __restrict__ gate, int n) {
  int i = blockIdx.x * blockDim.x + threadIdx.x;
  if (i < n) {
    float g = gate[i];
    attn[i] *= 1.f / (1.f + __expf(-g));
  }
}

// h = rms(o)*w_pa ; residual2 = h + residual1 ; h2 = rms(residual2)*w_pm
__global__ __launch_bounds__(256) void postattn_kernel(
    const float* __restrict__ o, const float* __restrict__ res1,
    const float* __restrict__ w_pa, const float* __restrict__ w_pm,
    float* __restrict__ res2_out, float* __restrict__ h2_out) {
  int t = blockIdx.x, tid = threadIdx.x;
  float ov[8], ss = 0.f;
#pragma unroll
  for (int i = 0; i < 8; ++i) {
    int c = tid + i * 256;
    float x = o[(size_t)t * kHid + c];
    ov[i] = x;
    ss += x * x;
  }
  ss = blockReduceSum(ss);
  float rs1 = rsqrtf(ss / kHid + kEps);
  float rv[8], ss2 = 0.f;
#pragma unroll
  for (int i = 0; i < 8; ++i) {
    int c = tid + i * 256;
    float h = ov[i] * rs1 * w_pa[c];
    float r = h + res1[(size_t)t * kHid + c];
    rv[i] = r;
    ss2 += r * r;
  }
  ss2 = blockReduceSum(ss2);
  float rs2 = rsqrtf(ss2 / kHid + kEps);
#pragma unroll
  for (int i = 0; i < 8; ++i) {
    int c = tid + i * 256;
    res2_out[(size_t)t * kHid + c] = rv[i];
    h2_out[(size_t)t * kHid + c] = rv[i] * rs2 * w_pm[c];
  }
}

// grouped top-k router; one thread per token
__global__ void router_kernel(const float* __restrict__ logits,
                              const float* __restrict__ bias,
                              float* __restrict__ combine) {
  int t = blockIdx.x * blockDim.x + threadIdx.x;
  if (t >= kT) return;
  float s[kE], b[kE];
#pragma unroll
  for (int e = 0; e < kE; ++e) {
    float l = logits[(size_t)t * kE + e];
    s[e] = 1.f / (1.f + __expf(-l));
    b[e] = s[e] + bias[e];
  }
  float gs[4];
#pragma unroll
  for (int g = 0; g < 4; ++g) {
    float m1 = -__builtin_inff(), m2 = -__builtin_inff();
    for (int k = 0; k < 8; ++k) {
      float x = b[g * 8 + k];
      if (x > m1) { m2 = m1; m1 = x; } else if (x > m2) { m2 = x; }
    }
    gs[g] = m1 + m2;
  }
  int g1 = 0;
  for (int g = 1; g < 4; ++g) if (gs[g] > gs[g1]) g1 = g;
  int g2 = -1;
  for (int g = 0; g < 4; ++g) {
    if (g == g1) continue;
    if (g2 < 0 || gs[g] > gs[g2]) g2 = g;
  }
  float mb[kE];
#pragma unroll
  for (int e = 0; e < kE; ++e) {
    int g = e >> 3;
    mb[e] = (g == g1 || g == g2) ? b[e] : -__builtin_inff();
  }
  int ids[4]; float tw[4]; float wsum = 0.f;
  for (int k = 0; k < 4; ++k) {
    int best = 0; float bv = -__builtin_inff();
    for (int e = 0; e < kE; ++e)
      if (mb[e] > bv) { bv = mb[e]; best = e; }
    ids[k] = best;
    mb[best] = -__builtin_inff();
    tw[k] = s[best];
    wsum += tw[k];
  }
  float out[kE];
#pragma unroll
  for (int e = 0; e < kE; ++e) out[e] = 0.f;
  for (int k = 0; k < 4; ++k) out[ids[k]] = tw[k] / wsum * kRouteScale;
#pragma unroll
  for (int e = 0; e < kE; ++e) combine[(size_t)t * kE + e] = out[e];
}

// MoE: for each 16-token tile, loop experts (skip inactive) + shared expert.
// stage1: a = silu(h@w1)*(h@w3) * combine -> LDS (f16); stage2: routed += a@w2.
// Strided weight tiles are gathered by the Tensor Data Mover into LDS.
__global__ __launch_bounds__(256) void moe_kernel(
    const float* __restrict__ h2, const float* __restrict__ w1e,
    const float* __restrict__ w3e, const float* __restrict__ w2e,
    const float* __restrict__ w1s, const float* __restrict__ w3s,
    const float* __restrict__ w2s, const float* __restrict__ combine,
    float* __restrict__ routed) {
  __shared__ _Float16 aT[16][kI];  // 16 KB
#if USE_TDM
  __shared__ float stg[8][2][32 * 16];  // per-wave TDM staging, 32 KB
#endif
  int lane = threadIdx.x & 31, wv = threadIdx.x >> 5;
  int hf = lane >> 4, col = lane & 15;
  int t0 = blockIdx.x * 16;

  for (int e = 0; e <= kE; ++e) {
    const float *W1, *W3, *W2;
    if (e < kE) {
      bool act = false;
      for (int r = 0; r < 16; ++r)
        if (combine[(size_t)(t0 + r) * kE + e] != 0.f) act = true;
      if (!act) continue;  // uniform across block
      W1 = w1e + (size_t)e * kHid * kI;
      W3 = w3e + (size_t)e * kHid * kI;
      W2 = w2e + (size_t)e * kI * kHid;
    } else {
      W1 = w1s; W3 = w3s; W2 = w2s;  // shared expert, weight 1
    }
    // stage 1: a tile [16, 512]
    for (int it = wv; it < kI / 16; it += 8) {
      int n0 = it * 16;
      v8f g = {}, u = {};
      for (int k0 = 0; k0 < kHid; k0 += 32) {
        v16h a = fragA_f32(h2, kHid, t0, k0, lane);
        v16h b1, b3;
#if USE_TDM
        tdm_load_tile_f32((unsigned)(size_t)&stg[wv][0][0],
                          W1 + (size_t)k0 * kI + n0, 16, 32, kI, kI, kHid);
        tdm_load_tile_f32((unsigned)(size_t)&stg[wv][1][0],
                          W3 + (size_t)k0 * kI + n0, 16, 32, kI, kI, kHid);
        __builtin_amdgcn_s_wait_tensorcnt(0);
        b1 = fragB_ldsTile(&stg[wv][0][0], lane);
        b3 = fragB_ldsTile(&stg[wv][1][0], lane);
#else
        b1 = fragB_f32_colK(W1, kI, n0, k0, lane);
        b3 = fragB_f32_colK(W3, kI, n0, k0, lane);
#endif
        g = wmma32(a, b1, g);
        u = wmma32(a, b3, u);
      }
#pragma unroll
      for (int r = 0; r < 8; ++r) {
        int row = r + hf * 8;
        float cw = (e < kE) ? combine[(size_t)(t0 + row) * kE + e] : 1.f;
        float gv = g[r];
        float av = (gv / (1.f + __expf(-gv))) * u[r] * cw;  // silu(g)*u
        aT[row][n0 + col] = (_Float16)av;
      }
    }
    __syncthreads();
    // stage 2: routed[16, 2048] += aT @ W2 ; two n-subtiles per A fragment
    for (int nt = wv * 2; nt < kHid / 16; nt += 16) {
      int n0 = nt * 16, n1 = n0 + 16;
      v8f acc0 = {}, acc1 = {};
      for (int k0 = 0; k0 < kI; k0 += 32) {
        v16h a = fragA_f16(&aT[col][k0 + hf * 16]);  // ds_load path
        v16h b0, b1;
#if USE_TDM
        tdm_load_tile_f32((unsigned)(size_t)&stg[wv][0][0],
                          W2 + (size_t)k0 * kHid + n0, 16, 32, kHid, kHid, kI);
        tdm_load_tile_f32((unsigned)(size_t)&stg[wv][1][0],
                          W2 + (size_t)k0 * kHid + n1, 16, 32, kHid, kHid, kI);
        __builtin_amdgcn_s_wait_tensorcnt(0);
        b0 = fragB_ldsTile(&stg[wv][0][0], lane);
        b1 = fragB_ldsTile(&stg[wv][1][0], lane);
#else
        b0 = fragB_f32_colK(W2, kHid, n0, k0, lane);
        b1 = fragB_f32_colK(W2, kHid, n1, k0, lane);
#endif
        acc0 = wmma32(a, b0, acc0);
        acc1 = wmma32(a, b1, acc1);
      }
#pragma unroll
      for (int r = 0; r < 8; ++r) {
        int row = r + hf * 8;
        routed[(size_t)(t0 + row) * kHid + n0 + col] += acc0[r];
        routed[(size_t)(t0 + row) * kHid + n1 + col] += acc1[r];
      }
    }
    __syncthreads();
  }
}

// out = rms(x) * w
__global__ __launch_bounds__(256) void final_norm_kernel(
    const float* __restrict__ x, const float* __restrict__ w,
    float* __restrict__ out) {
  int t = blockIdx.x, tid = threadIdx.x;
  float v[8], ss = 0.f;
#pragma unroll
  for (int i = 0; i < 8; ++i) {
    int c = tid + i * 256;
    float y = x[(size_t)t * kHid + c];
    v[i] = y;
    ss += y * y;
  }
  ss = blockReduceSum(ss);
  float rs = rsqrtf(ss / kHid + kEps);
#pragma unroll
  for (int i = 0; i < 8; ++i) {
    int c = tid + i * 256;
    out[(size_t)t * kHid + c] = v[i] * rs * w[c];
  }
}

// ------------------------- host orchestration -------------------------------

extern "C" void kernel_launch(void* const* d_in, const int* in_sizes, int n_in,
                              void* d_out, int out_size, void* d_ws,
                              size_t ws_size, hipStream_t stream) {
  (void)in_sizes; (void)n_in; (void)out_size; (void)ws_size;

  const int*   positions = (const int*)d_in[0];
  const float* hs        = (const float*)d_in[1];
  const float* res_in    = (const float*)d_in[2];
  const float* w_in_ln   = (const float*)d_in[3];
  const float* w_pa_ln   = (const float*)d_in[4];
  const float* w_pm_ln   = (const float*)d_in[5];
  const float* w_po_ln   = (const float*)d_in[6];
  const float* w_qn      = (const float*)d_in[7];
  const float* w_kn      = (const float*)d_in[8];
  const float* w_qkv     = (const float*)d_in[9];
  const float* w_gate    = (const float*)d_in[10];
  const float* w_o       = (const float*)d_in[11];
  const float* w_router  = (const float*)d_in[12];
  const float* ebias     = (const float*)d_in[13];
  const float* w1e       = (const float*)d_in[14];
  const float* w3e       = (const float*)d_in[15];
  const float* w2e       = (const float*)d_in[16];
  const float* w1s       = (const float*)d_in[17];
  const float* w3s       = (const float*)d_in[18];
  const float* w2s       = (const float*)d_in[19];

  float* out_h   = (float*)d_out;
  float* out_res = out_h + (size_t)kT * kHid;

  // bump allocator over d_ws
  char* wsp = (char*)d_ws;
  auto alloc = [&](size_t bytes) -> char* {
    char* p = wsp;
    wsp += (bytes + 255) & ~(size_t)255;
    return p;
  };
  float*    h1     = (float*)alloc(sizeof(float) * kT * kHid);
  float*    res1   = (float*)alloc(sizeof(float) * kT * kHid);
  float*    qkv    = (float*)alloc(sizeof(float) * kT * kQKV);
  float*    gatebf = (float*)alloc(sizeof(float) * kT * kQ);
  _Float16* q16    = (_Float16*)alloc(sizeof(_Float16) * kT * kNH * kHD);
  _Float16* k16    = (_Float16*)alloc(sizeof(_Float16) * kT * kNKV * kHD);
  _Float16* vT     = (_Float16*)alloc(sizeof(_Float16) * kNKV * kHD * kT + 256);
  float*    attn   = (float*)alloc(sizeof(float) * kT * kQ);
  float*    oout   = (float*)alloc(sizeof(float) * kT * kHid);
  float*    h2     = (float*)alloc(sizeof(float) * kT * kHid);
  float*    logits = (float*)alloc(sizeof(float) * kT * kE);
  float*    comb   = (float*)alloc(sizeof(float) * kT * kE);
  float*    routed = (float*)alloc(sizeof(float) * kT * kHid);

  auto gx = [](int n) { return (n + 255) / 256; };

  // 1. residual add + input RMSNorm
  add_norm_kernel<<<kT, 256, 0, stream>>>(hs, res_in, w_in_ln, res1, h1);
  // 2. qkv = h1 @ w_qkv^T ; 3. gate = h1 @ w_gate^T
  gemm_nt_kernel<<<dim3(gx(kQKV), kT / 16), 128, 0, stream>>>(
      h1, w_qkv, qkv, kT, kQKV, kHid);
  gemm_nt_kernel<<<dim3(gx(kQ), kT / 16), 128, 0, stream>>>(
      h1, w_gate, gatebf, kT, kQ, kHid);
  // 4. per-head RMS + RoPE, v transpose to f16
  qkv_prep_kernel<<<kT, 128, 0, stream>>>(qkv, positions, w_qn, w_kn, q16, k16,
                                          vT);
  // 5. sliding-window flash attention
  attn_kernel<<<dim3(kT / 16, kNH), 32, 0, stream>>>(q16, k16, vT, attn);
  // 6. attn *= sigmoid(gate)
  gate_kernel<<<(kT * kQ + 255) / 256, 256, 0, stream>>>(attn, gatebf,
                                                         kT * kQ);
  // 7. o-proj
  gemm_nt_kernel<<<dim3(gx(kHid), kT / 16), 128, 0, stream>>>(
      attn, w_o, oout, kT, kHid, kQ);
  // 8. post-attn norm + residual + pre-mlp norm
  postattn_kernel<<<kT, 256, 0, stream>>>(oout, res1, w_pa_ln, w_pm_ln,
                                          out_res, h2);
  // 9. router logits
  gemm_nt_kernel<<<dim3(gx(kE), kT / 16), 128, 0, stream>>>(
      h2, w_router, logits, kT, kE, kHid);
  // 10. grouped top-k routing -> dense combine [T, E]
  router_kernel<<<(kT + 255) / 256, 256, 0, stream>>>(logits, ebias, comb);
  // 11. MoE (routed experts with tile skipping + shared expert)
  hipMemsetAsync(routed, 0, sizeof(float) * kT * kHid, stream);
  moe_kernel<<<kT / 16, 256, 0, stream>>>(h2, w1e, w3e, w2e, w1s, w3s, w2s,
                                          comb, routed);
  // 12. final RMSNorm
  final_norm_kernel<<<kT, 256, 0, stream>>>(routed, w_po_ln, out_h);
}
